// Attention_27573690040550
// MI455X (gfx1250) — matrix operand order repaired and checked by reading, not compiled, not run on hardware
//
#include <hip/hip_runtime.h>
#include <hip/hip_bf16.h>
#include <math.h>

// ---------------------------------------------------------------------------
// B=2, T=2048, C=1024, H=16 (d=64).  All matmuls on v_wmma_f32_16x16x32_f16.
// GEMMs: 64x64 tile per wave (16 accumulators) -> 1 b128 load per WMMA.
// Attention: flash-style, K/V tiles staged to LDS by the Tensor Data Mover
// (tensor_load_to_lds, double-buffered, s_wait_tensorcnt) on gfx1250.
// ---------------------------------------------------------------------------

typedef __attribute__((ext_vector_type(16))) _Float16 v16h;
typedef __attribute__((ext_vector_type(8)))  _Float16 v8h;
typedef __attribute__((ext_vector_type(8)))  float    v8f;
typedef __attribute__((ext_vector_type(4)))  unsigned int v4u;
typedef __attribute__((ext_vector_type(8)))  int      v8i;
typedef __attribute__((ext_vector_type(4)))  int      v4i;

#define WMMA_F16(a, b, c) \
    __builtin_amdgcn_wmma_f32_16x16x32_f16(false, (a), false, (b), (short)0, (c), false, false)

// A-fragment (16x32 f16, row-major source): lane holds row M = lane%16.
// Element i: K = hl*8 + i (i<8); K = 16 + hl*8 + (i-8) (i>=8).
static __device__ __forceinline__ v16h load_a_frag(const _Float16* row_k0, int hl) {
    v8h c0 = *(const v8h*)(row_k0 + hl * 8);
    v8h c1 = *(const v8h*)(row_k0 + 16 + hl * 8);
    return __builtin_shufflevector(c0, c1, 0, 1, 2, 3, 4, 5, 6, 7, 8, 9, 10, 11, 12, 13, 14, 15);
}

// B-fragment (32x16 f16) from K-contiguous source: lane = column N,
// el i: K = hl*8 + i (i<8); K = 16 + hl*8 + (i-8) (i>=8).
static __device__ __forceinline__ v16h load_b_frag(const _Float16* col_k0, int hl) {
    v8h c0 = *(const v8h*)(col_k0 + hl * 8);
    v8h c1 = *(const v8h*)(col_k0 + 16 + hl * 8);
    return __builtin_shufflevector(c0, c1, 0, 1, 2, 3, 4, 5, 6, 7, 8, 9, 10, 11, 12, 13, 14, 15);
}

// ---------------------------------------------------------------------------
// Tensor Data Mover: 2D tile load Global->LDS.  Builds a D# descriptor:
//   group0: count=1 | lds_addr | global_addr[56:0] | type=2
//   group1: data_size=2B, tensor_dim0/1, tile_dim0/1, tensor_dim0_stride
// Dims/strides are in elements (data_size units); addresses in bytes.
// ---------------------------------------------------------------------------
static __device__ __forceinline__ void tdm_load_2d(
    unsigned int lds_off, const _Float16* gptr,
    unsigned int tile_d0, unsigned int tile_d1,
    unsigned int tensor_d0, unsigned int tensor_d1, unsigned int stride0)
{
    unsigned long long ga = (unsigned long long)(uintptr_t)gptr;
    v4u g0;
    g0[0] = 1u;                                        // count=1, user mode
    g0[1] = lds_off;                                   // lds_addr (bytes)
    g0[2] = (unsigned int)ga;                          // global_addr[31:0]
    g0[3] = (unsigned int)((ga >> 32) & 0x1FFFFFFu)    // global_addr[56:32]
          | (2u << 30);                                // type=2 ("image")
    v8i g1;
    g1[0] = (int)(1u << 16);                           // data_size=1 -> 2 bytes
    g1[1] = (int)((tensor_d0 & 0xFFFFu) << 16);        // tensor_dim0[15:0]
    g1[2] = (int)((tensor_d0 >> 16) | ((tensor_d1 & 0xFFFFu) << 16));
    g1[3] = (int)((tensor_d1 >> 16) | (tile_d0 << 16));// tile_dim0
    g1[4] = (int)(tile_d1 & 0xFFFFu);                  // tile_dim1 (tile_dim2=0)
    g1[5] = (int)stride0;                              // tensor_dim0_stride[31:0]
    g1[6] = 0;
    g1[7] = 0;
    v4i z4 = {0, 0, 0, 0};
#if __clang_major__ >= 23
    v8i z8 = {0, 0, 0, 0, 0, 0, 0, 0};
    __builtin_amdgcn_tensor_load_to_lds(g0, g1, z4, z4, z8, 0);
#else
    __builtin_amdgcn_tensor_load_to_lds(g0, g1, z4, z4, 0);
#endif
}

// ---------------------------------------------------------------------------
// Stage 0: fp32 -> f16 conversions / transposes
// ---------------------------------------------------------------------------
__global__ void k_cvt_x(const float* __restrict__ in, _Float16* __restrict__ out) {
    size_t i = (size_t)blockIdx.x * 256 + threadIdx.x;
    out[i] = (_Float16)in[i];
}
__global__ void k_tr_wqkv(const float* __restrict__ w, _Float16* __restrict__ out) {
    size_t i = (size_t)blockIdx.x * 256 + threadIdx.x;   // over 3072*1024
    int n = (int)(i >> 10), k = (int)(i & 1023);
    out[i] = (_Float16)w[(size_t)k * 3072 + n];
}
__global__ void k_tr_wo(const float* __restrict__ w, _Float16* __restrict__ out) {
    size_t i = (size_t)blockIdx.x * 256 + threadIdx.x;   // over 1024*1024
    int n = (int)(i >> 10), k = (int)(i & 1023);
    out[i] = (_Float16)w[(size_t)k * 1024 + n];
}

// ---------------------------------------------------------------------------
// Stage 1: QKV GEMM [4096 x 3072], 64x64 tile per wave (16 WMMA / K-step)
// ---------------------------------------------------------------------------
__global__ __launch_bounds__(128) void k_gemm_qkv(
    const _Float16* __restrict__ xh,     // [4096][1024]
    const _Float16* __restrict__ wT,     // [3072][1024]
    const float*    __restrict__ bias,   // [3072]
    _Float16* __restrict__ qh,           // [32][2048][64]
    _Float16* __restrict__ kh,           // [32][2048][64]
    _Float16* __restrict__ vT)           // [32][64][2048]
{
    const int lane = threadIdx.x & 31;
    const int wid  = threadIdx.x >> 5;
    const int hl   = lane >> 4;
    const int l16  = lane & 15;
    const int tn   = blockIdx.x * 4 + wid;   // 64-col block [0,48)
    const int tm   = blockIdx.y;             // 64-row block [0,64)

    const _Float16* arow[4];
    const _Float16* brow[4];
#pragma unroll
    for (int i = 0; i < 4; ++i) {
        arow[i] = xh + (size_t)(tm * 64 + i * 16 + l16) * 1024;
        brow[i] = wT + (size_t)(tn * 64 + i * 16 + l16) * 1024;
    }

    v8f acc[4][4];
#pragma unroll
    for (int mi = 0; mi < 4; ++mi)
#pragma unroll
        for (int ni = 0; ni < 4; ++ni) acc[mi][ni] = (v8f){};

#pragma unroll 2
    for (int k0 = 0; k0 < 1024; k0 += 32) {
        v16h a[4], b[4];
#pragma unroll
        for (int i = 0; i < 4; ++i) a[i] = load_a_frag(arow[i] + k0, hl);
#pragma unroll
        for (int i = 0; i < 4; ++i) b[i] = load_b_frag(brow[i] + k0, hl);
#pragma unroll
        for (int mi = 0; mi < 4; ++mi)
#pragma unroll
            for (int ni = 0; ni < 4; ++ni)
                acc[mi][ni] = WMMA_F16(a[mi], b[ni], acc[mi][ni]);
    }

#pragma unroll
    for (int ni = 0; ni < 4; ++ni) {
        const int n_global = tn * 64 + ni * 16 + l16;  // [0,3072)
        const float bn  = bias[n_global];
        const int which = n_global >> 10;              // 0=Q,1=K,2=V
        const int c     = n_global & 1023;
        const int head  = c >> 6;
        const int dd    = c & 63;
#pragma unroll
        for (int mi = 0; mi < 4; ++mi) {
#pragma unroll
            for (int r = 0; r < 8; ++r) {
                int m_global = tm * 64 + mi * 16 + r + 8 * hl;
                int bb = m_global >> 11;
                int t  = m_global & 2047;
                int bh = bb * 16 + head;
                _Float16 v = (_Float16)(acc[mi][ni][r] + bn);
                if (which == 0)      qh[(size_t)bh * 131072 + (size_t)t * 64 + dd] = v;
                else if (which == 1) kh[(size_t)bh * 131072 + (size_t)t * 64 + dd] = v;
                else                 vT[(size_t)bh * 131072 + (size_t)dd * 2048 + t] = v;
            }
        }
    }
}

// ---------------------------------------------------------------------------
// Stage 2: causal flash attention.  2 waves/block; each wave owns a 16-query
// tile and double-buffered LDS K/V tiles filled by the Tensor Data Mover.
// ---------------------------------------------------------------------------
__global__ __launch_bounds__(64) void k_attn(
    const _Float16* __restrict__ qh,
    const _Float16* __restrict__ kh,
    const _Float16* __restrict__ vT,
    _Float16* __restrict__ ao)           // [4096][1024]
{
    __shared__ __align__(128) _Float16 ldsK[2][2][32 * 64];  // [wave][buf] 4KB
    __shared__ __align__(128) _Float16 ldsV[2][2][64 * 32];  // [wave][buf] 4KB
    __shared__ __align__(32)  _Float16 ldsP[2][16 * 32];     // P relayout

    const int lane = threadIdx.x & 31;
    const int hl   = lane >> 4;
    const int l16  = lane & 15;
    const int swid = __builtin_amdgcn_readfirstlane(threadIdx.x >> 5); // wave-scalar
    const int qt   = blockIdx.x * 2 + swid;   // [0,128)
    const int bh   = blockIdx.y;              // [0,32)

    const _Float16* qb = qh + (size_t)bh * 131072;
    const _Float16* kb = kh + (size_t)bh * 131072;
    const _Float16* vb = vT + (size_t)bh * 131072;

    const unsigned int ldsK_off[2] = {
        (unsigned int)(uintptr_t)&ldsK[swid][0][0],
        (unsigned int)(uintptr_t)&ldsK[swid][1][0]};
    const unsigned int ldsV_off[2] = {
        (unsigned int)(uintptr_t)&ldsV[swid][0][0],
        (unsigned int)(uintptr_t)&ldsV[swid][1][0]};

    // Preload Q A-fragments (d=64 -> two 32-wide K steps).
    const _Float16* qrow = qb + (size_t)(qt * 16 + l16) * 64;
    v16h aq0 = load_a_frag(qrow + 0, hl);
    v16h aq1 = load_a_frag(qrow + 32, hl);

    v8f o[4] = {{}, {}, {}, {}};
    float mrow[8], lrow[8];
#pragma unroll
    for (int r = 0; r < 8; ++r) { mrow[r] = -__builtin_inff(); lrow[r] = 0.0f; }

    const float scale = 0.03125f;            // 1/sqrt(C)
    const int kc_max = (qt * 16 + 15) >> 5;
    _Float16* P = ldsP[swid];

    // TDM: stage chunk 0 (K tile: 32 keys x 64, row-major; V tile from vT:
    // 64 dd-rows x 32 t-cols) into buffer 0.
    tdm_load_2d(ldsK_off[0], kb, /*tile*/64, 32, /*tensor*/64, 2048, /*stride0*/64);
    tdm_load_2d(ldsV_off[0], vb, /*tile*/32, 64, /*tensor*/2048, 64, /*stride0*/2048);

    for (int kc = 0; kc <= kc_max; ++kc) {
        const int buf = kc & 1;
        if (kc < kc_max) {   // prefetch next chunk into the other buffer
            tdm_load_2d(ldsK_off[buf ^ 1], kb + (size_t)(kc + 1) * 32 * 64,
                        64, 32, 64, 2048, 64);
            tdm_load_2d(ldsV_off[buf ^ 1], vb + (size_t)(kc + 1) * 32,
                        32, 64, 2048, 64, 2048);
            __builtin_amdgcn_s_wait_tensorcnt(2);   // current chunk landed
        } else {
            __builtin_amdgcn_s_wait_tensorcnt(0);
        }
        asm volatile("" ::: "memory");   // TDM wrote LDS behind compiler's back

        const _Float16* Kt = &ldsK[swid][buf][0];   // [32 rows][64]
        const _Float16* Vt = &ldsV[swid][buf][0];   // [64 rows][32]

        // ---- scores for keys kc*32 .. kc*32+31 (two 16-key tiles)
        v8f s0 = {}, s1 = {};
        {
            const _Float16* kr = Kt + (size_t)l16 * 64;
            s0 = WMMA_F16(aq0, load_b_frag(kr + 0, hl), s0);
            s0 = WMMA_F16(aq1, load_b_frag(kr + 32, hl), s0);
        }
        {
            const _Float16* kr = Kt + (size_t)(16 + l16) * 64;
            s1 = WMMA_F16(aq0, load_b_frag(kr + 0, hl), s1);
            s1 = WMMA_F16(aq1, load_b_frag(kr + 32, hl), s1);
        }

        // ---- mask + online softmax (row r lives in one 16-lane half)
#pragma unroll
        for (int r = 0; r < 8; ++r) {
            int tq  = qt * 16 + r + 8 * hl;
            int tk0 = kc * 32 + l16;
            int tk1 = tk0 + 16;
            float x0 = (tk0 <= tq) ? s0[r] * scale : -__builtin_inff();
            float x1 = (tk1 <= tq) ? s1[r] * scale : -__builtin_inff();

            float mx = fmaxf(x0, x1);
#pragma unroll
            for (int off = 1; off < 16; off <<= 1)
                mx = fmaxf(mx, __shfl_xor(mx, off, 32));

            float mnew  = fmaxf(mrow[r], mx);
            float alpha = __expf(mrow[r] - mnew);
            float p0 = __expf(x0 - mnew);
            float p1 = __expf(x1 - mnew);
            float ps = p0 + p1;
#pragma unroll
            for (int off = 1; off < 16; off <<= 1)
                ps += __shfl_xor(ps, off, 32);

            lrow[r] = lrow[r] * alpha + ps;
            mrow[r] = mnew;
            o[0][r] *= alpha; o[1][r] *= alpha; o[2][r] *= alpha; o[3][r] *= alpha;

            P[(r + 8 * hl) * 32 + l16]      = (_Float16)p0;
            P[(r + 8 * hl) * 32 + 16 + l16] = (_Float16)p1;
        }
        asm volatile("" ::: "memory");   // per-wave LDS ops are in-order

        // ---- O += P * V  (A = P 16x32 from LDS, B rows = Vt[dd][t-chunk])
        v16h ap = load_a_frag(P + l16 * 32, hl);
#pragma unroll
        for (int nn = 0; nn < 4; ++nn) {
            const _Float16* vr = Vt + (size_t)(nn * 16 + l16) * 32;
            o[nn] = WMMA_F16(ap, load_b_frag(vr, hl), o[nn]);
        }
        asm volatile("" ::: "memory");
    }

    // ---- normalize + store f16 attention output, row-major [b,t,c]
    const int b = bh >> 4, h = bh & 15;
#pragma unroll
    for (int r = 0; r < 8; ++r) {
        int tq = qt * 16 + r + 8 * hl;
        float inv = 1.0f / lrow[r];
        size_t rowbase = ((size_t)(b * 2048 + tq)) * 1024 + h * 64;
#pragma unroll
        for (int nn = 0; nn < 4; ++nn)
            ao[rowbase + nn * 16 + l16] = (_Float16)(o[nn][r] * inv);
    }
}

// ---------------------------------------------------------------------------
// Stage 3: output projection [4096 x 1024], 64x64 tile per wave, fp32 out
// ---------------------------------------------------------------------------
__global__ __launch_bounds__(128) void k_gemm_out(
    const _Float16* __restrict__ ah,     // [4096][1024]
    const _Float16* __restrict__ wT,     // [1024][1024]
    const float*    __restrict__ bias,   // [1024]
    float* __restrict__ out)             // [4096][1024]
{
    const int lane = threadIdx.x & 31;
    const int wid  = threadIdx.x >> 5;
    const int hl   = lane >> 4;
    const int l16  = lane & 15;
    const int tn   = blockIdx.x * 4 + wid;   // [0,16)
    const int tm   = blockIdx.y;             // [0,64)

    const _Float16* arow[4];
    const _Float16* brow[4];
#pragma unroll
    for (int i = 0; i < 4; ++i) {
        arow[i] = ah + (size_t)(tm * 64 + i * 16 + l16) * 1024;
        brow[i] = wT + (size_t)(tn * 64 + i * 16 + l16) * 1024;
    }

    v8f acc[4][4];
#pragma unroll
    for (int mi = 0; mi < 4; ++mi)
#pragma unroll
        for (int ni = 0; ni < 4; ++ni) acc[mi][ni] = (v8f){};

#pragma unroll 2
    for (int k0 = 0; k0 < 1024; k0 += 32) {
        v16h a[4], b[4];
#pragma unroll
        for (int i = 0; i < 4; ++i) a[i] = load_a_frag(arow[i] + k0, hl);
#pragma unroll
        for (int i = 0; i < 4; ++i) b[i] = load_b_frag(brow[i] + k0, hl);
#pragma unroll
        for (int mi = 0; mi < 4; ++mi)
#pragma unroll
            for (int ni = 0; ni < 4; ++ni)
                acc[mi][ni] = WMMA_F16(a[mi], b[ni], acc[mi][ni]);
    }

#pragma unroll
    for (int ni = 0; ni < 4; ++ni) {
        const int n_global = tn * 64 + ni * 16 + l16;
        const float bn = bias[n_global];
#pragma unroll
        for (int mi = 0; mi < 4; ++mi) {
#pragma unroll
            for (int r = 0; r < 8; ++r) {
                int m_global = tm * 64 + mi * 16 + r + 8 * hl;
                out[(size_t)m_global * 1024 + n_global] = acc[mi][ni][r] + bn;
            }
        }
    }
}

// ---------------------------------------------------------------------------
extern "C" void kernel_launch(void* const* d_in, const int* in_sizes, int n_in,
                              void* d_out, int out_size, void* d_ws, size_t ws_size,
                              hipStream_t stream) {
    const float* x    = (const float*)d_in[0];   // [2,2048,1024]
    const float* Wqkv = (const float*)d_in[1];   // [1024,3072]
    const float* bqkv = (const float*)d_in[2];   // [3072]
    const float* Wo   = (const float*)d_in[3];   // [1024,1024]
    const float* bo   = (const float*)d_in[4];   // [1024]
    float* out = (float*)d_out;                  // [2,2048,1024]

    _Float16* xh    = (_Float16*)d_ws;               // 4096*1024
    _Float16* wqkvT = xh    + (size_t)4096 * 1024;   // 3072*1024
    _Float16* woT   = wqkvT + (size_t)3072 * 1024;   // 1024*1024
    _Float16* qh    = woT   + (size_t)1024 * 1024;   // 32*2048*64
    _Float16* kh    = qh    + (size_t)32 * 131072;
    _Float16* vT    = kh    + (size_t)32 * 131072;
    _Float16* ao    = vT    + (size_t)32 * 131072;   // 4096*1024

    k_cvt_x  <<<dim3((4096 * 1024) / 256), 256, 0, stream>>>(x, xh);
    k_tr_wqkv<<<dim3((3072 * 1024) / 256), 256, 0, stream>>>(Wqkv, wqkvT);
    k_tr_wo  <<<dim3((1024 * 1024) / 256), 256, 0, stream>>>(Wo, woT);

    k_gemm_qkv<<<dim3(12, 64), 128, 0, stream>>>(xh, wqkvT, bqkv, qh, kh, vT);
    k_attn    <<<dim3(64, 32), 64,  0, stream>>>(qh, kh, vT, ao);
    k_gemm_out<<<dim3(4, 64),  128, 0, stream>>>(ao, woT, bo, out);
}
